// EngramModule_81862076662198
// MI455X (gfx1250) — matrix operand order repaired and checked by reading, not compiled, not run on hardware
//
#include <hip/hip_runtime.h>
#include <stdint.h>

// ---------------------------------------------------------------------------
// EngramModule fused kernel for MI455X (gfx1250, wave32, WMMA bf16)
// ---------------------------------------------------------------------------
typedef __attribute__((ext_vector_type(8)))  __bf16          v8bf;
typedef __attribute__((ext_vector_type(16))) __bf16          v16bf;
typedef __attribute__((ext_vector_type(8)))  float           v8f;
typedef __attribute__((ext_vector_type(8)))  unsigned short  u16x8;

constexpr int B_   = 4;
constexpr int L_   = 4096;
constexpr int HC   = 2;
constexpr int D_   = 1024;
constexpr int NH   = 8;
constexpr int ED   = 64;
constexpr int TE   = 512;     // TOTAL_ENGRAM = NH*ED (GEMM K)
constexpr int TM   = 45;      // tokens owned per block
constexpr int MC   = 48;      // computed rows: 3-token causal halo + 45, = 3 WMMA M tiles
constexpr int APAD = 520;     // bf16 A-tile row stride (pad: banks differ per row)
constexpr int KVP  = 1028;    // f32 KV row stride (pad: kill lane/lane+16 conflicts)
constexpr int NCHUNK = (L_ + TM - 1) / TM;   // 92

struct SMem {
  alignas(16) unsigned short A[MC][APAD];   // gathered mem rows, bf16     (~48.8 KB)
  alignas(16) float          KV[MC][KVP];   // k0 / k1 / value, reused     (~192.8 KB)
  alignas(16) float4         convw[HC*D_];  // depthwise taps               (32 KB)
  float                      convb[HC*D_];  //                              (8 KB)
  float                      vb[D_];        // value bias                   (4 KB)
  float                      gate[MC][HC];  // sigmoid gates
};                                          // total ~286 KB < 320 KB/WGP

__device__ __forceinline__ unsigned short f2bf(float f) {
  unsigned int u = __float_as_uint(f);
  u += 0x7FFFu + ((u >> 16) & 1u);          // round-to-nearest-even
  return (unsigned short)(u >> 16);
}

__device__ __forceinline__ v16bf cat16(v8bf lo, v8bf hi) {
  return __builtin_shufflevector(lo, hi, 0,1,2,3,4,5,6,7,8,9,10,11,12,13,14,15);
}

// 48x1024 += A(48x512,bf16 LDS) * Wt(1024x512,bf16 global row-major [n][k])
// Wave layout: 8 waves stride the 64 N-tiles; each wave holds 3 M-tile accums
// and streams one B K-chunk per step -> 3 v_wmma_f32_16x16x32_bf16 per chunk.
__device__ __forceinline__ void gemm512(const unsigned short* __restrict__ wT,
                                        SMem* sm, int lane, int wave) {
  const int laneN = lane & 15;
  const int laneG = lane >> 4;
  for (int nt = wave; nt < D_ / 16; nt += 8) {
    v8f acc0 = {}, acc1 = {}, acc2 = {};
    const unsigned short* Bp = wT + (size_t)(nt * 16 + laneN) * TE + laneG * 8;
    const unsigned short* Ap = &sm->A[laneN][laneG * 8];
#pragma unroll
    for (int kc = 0; kc < TE / 32; ++kc) {
      // B tile: lane = column n, K = kc*32 + g*8 + {0..7} and +16
      v8bf blo = *(const v8bf*)(Bp + kc * 32);
      v8bf bhi = *(const v8bf*)(Bp + kc * 32 + 16);
      v16bf bv = cat16(blo, bhi);
      // A tiles: lane = row m, K = kc*32 + g*8 + {0..7} and +16
      v8bf a0l = *(const v8bf*)(Ap + kc * 32);
      v8bf a0h = *(const v8bf*)(Ap + kc * 32 + 16);
      acc0 = __builtin_amdgcn_wmma_f32_16x16x32_bf16(false, cat16(a0l, a0h),
                                                     false, bv, (short)0, acc0,
                                                     false, false);
      v8bf a1l = *(const v8bf*)(Ap + 16 * APAD + kc * 32);
      v8bf a1h = *(const v8bf*)(Ap + 16 * APAD + kc * 32 + 16);
      acc1 = __builtin_amdgcn_wmma_f32_16x16x32_bf16(false, cat16(a1l, a1h),
                                                     false, bv, (short)0, acc1,
                                                     false, false);
      v8bf a2l = *(const v8bf*)(Ap + 32 * APAD + kc * 32);
      v8bf a2h = *(const v8bf*)(Ap + 32 * APAD + kc * 32 + 16);
      acc2 = __builtin_amdgcn_wmma_f32_16x16x32_bf16(false, cat16(a2l, a2h),
                                                     false, bv, (short)0, acc2,
                                                     false, false);
    }
    // C layout: lanes 0-15 -> M 0..7, lanes 16-31 -> M 8..15; col = lane%16
    const int col = nt * 16 + laneN;
#pragma unroll
    for (int j = 0; j < 8; ++j) {
      int row = laneG * 8 + j;
      sm->KV[row][col]      = acc0[j];
      sm->KV[16 + row][col] = acc1[j];
      sm->KV[32 + row][col] = acc2[j];
    }
  }
}

// RMSNorm(q) . RMSNorm(k) gate: s1=sum k^2, s2=sum q^2, s3=sum q*nqw*nkw*k
__device__ __forceinline__ void reduce_gate(int hc, SMem* sm,
                                            const float* __restrict__ hidden,
                                            const float* __restrict__ key_b,
                                            const float* __restrict__ nkw,
                                            const float* __restrict__ nqw,
                                            int b, int l0, int lane, int wave) {
  for (int r = 0; r < MC / 8; ++r) {           // 6 rows per wave
    const int i = wave * (MC / 8) + r;
    const int l = l0 - 3 + i;
    const bool valid = (l >= 0) && (l < L_);
    float s1 = 0.f, s2 = 0.f, s3 = 0.f;
    if (valid) {
      const float* qp = hidden + (((size_t)b * L_ + l) * HC + hc) * D_;
      const float* kb = key_b + hc * D_;
      const float* nk = nkw + hc * D_;
      const float* nq = nqw + hc * D_;
      for (int t = 0; t < D_ / 32; ++t) {
        int d = lane + t * 32;
        float kv = sm->KV[i][d] + kb[d];
        float q  = qp[d];
        s1 = fmaf(kv, kv, s1);
        s2 = fmaf(q, q, s2);
        s3 = fmaf(q * nq[d] * nk[d], kv, s3);
      }
    }
    for (int off = 16; off > 0; off >>= 1) {
      s1 += __shfl_xor(s1, off, 32);
      s2 += __shfl_xor(s2, off, 32);
      s3 += __shfl_xor(s3, off, 32);
    }
    if (lane == 0) {
      float g = 0.f;                           // l<0 halo: conv sees zeros
      if (valid) {
        float rk = rsqrtf(s1 * (1.f / 1024.f) + 1e-6f);
        float rq = rsqrtf(s2 * (1.f / 1024.f) + 1e-6f);
        float score = s3 * rk * rq * (1.f / 32.f);   // /sqrt(D)
        float aa = fabsf(score);
        float sq = sqrtf(fmaxf(aa, 1e-6f));
        float sg = score > 0.f ? sq : (score < 0.f ? -sq : 0.f);
        g = 1.f / (1.f + __expf(-sg));
      }
      sm->gate[i][hc] = g;
    }
  }
}

// Weight prep: f32 [K=512][N=1024] -> bf16 [N=1024][K=512], for value_w and
// both key_w heads. 3 MB into d_ws; L2-resident for the whole main kernel.
__global__ void prep_weights(const float* __restrict__ value_w,
                             const float* __restrict__ key_w,
                             unsigned short* __restrict__ wsB) {
  __shared__ float tile[32][33];
  const int m  = blockIdx.z;                   // 0=value, 1=key0, 2=key1
  const float* src = (m == 0) ? value_w : key_w + (size_t)(m - 1) * TE * D_;
  const int d0 = blockIdx.x * 32;              // N dim
  const int e0 = blockIdx.y * 32;              // K dim
  const int tx = threadIdx.x, ty = threadIdx.y;
  for (int j = ty; j < 32; j += 8)
    tile[j][tx] = src[(size_t)(e0 + j) * D_ + d0 + tx];
  __syncthreads();
  unsigned short* dst = wsB + (size_t)m * D_ * TE;
  for (int j = ty; j < 32; j += 8)
    dst[(size_t)(d0 + j) * TE + e0 + tx] = f2bf(tile[tx][j]);
}

__global__ __launch_bounds__(256, 1)
void engram_main(const float* __restrict__ hidden,
                 const int*   __restrict__ hash_ids,
                 const int*   __restrict__ offsets,
                 const float* __restrict__ emb,
                 const float* __restrict__ value_b,
                 const float* __restrict__ key_b,
                 const float* __restrict__ nkw,
                 const float* __restrict__ nqw,
                 const float* __restrict__ conv_w,
                 const float* __restrict__ conv_b,
                 const unsigned short* __restrict__ wsB,
                 float* __restrict__ out) {
  __shared__ SMem sm;
  const int tid  = threadIdx.x;
  const int lane = tid & 31;
  const int wave = tid >> 5;
  const int b    = blockIdx.y;
  const int l0   = blockIdx.x * TM;

  // ---- preload small per-channel tables --------------------------------
  for (int c = tid; c < HC * D_; c += 256) {
    sm.convw[c] = ((const float4*)conv_w)[c];
    sm.convb[c] = conv_b[c];
  }
  for (int d = tid; d < D_; d += 256) sm.vb[d] = value_b[d];

  // ---- gather: 48 rows x 8 heads x 64 f32 -> bf16 A tile ---------------
  {
    const int sub = tid & 7;                   // 8 lanes per (row, head)
    for (int p = tid >> 3; p < MC * NH; p += 32) {
      int i = p >> 3, h = p & 7;
      int l = l0 - 3 + i;
      int lg = l < 0 ? 0 : (l >= L_ ? L_ - 1 : l);
      int id = hash_ids[((size_t)b * L_ + lg) * NH + h] + offsets[h];
      const float4* src = (const float4*)(emb + (size_t)id * ED) + sub * 2;
      float4 x0 = src[0], x1 = src[1];
      u16x8 v;
      v[0] = f2bf(x0.x); v[1] = f2bf(x0.y); v[2] = f2bf(x0.z); v[3] = f2bf(x0.w);
      v[4] = f2bf(x1.x); v[5] = f2bf(x1.y); v[6] = f2bf(x1.z); v[7] = f2bf(x1.w);
      *(u16x8*)&sm.A[i][h * ED + sub * 8] = v;
    }
  }
  __syncthreads();

  // ---- k(hc=0) GEMM -> gates, k(hc=1) GEMM -> gates, value GEMM --------
  gemm512(wsB + (size_t)1 * D_ * TE, &sm, lane, wave);
  __syncthreads();
  reduce_gate(0, &sm, hidden, key_b, nkw, nqw, b, l0, lane, wave);
  __syncthreads();
  gemm512(wsB + (size_t)2 * D_ * TE, &sm, lane, wave);
  __syncthreads();
  reduce_gate(1, &sm, hidden, key_b, nkw, nqw, b, l0, lane, wave);
  __syncthreads();
  gemm512(wsB, &sm, lane, wave);               // value projection
  __syncthreads();

  // ---- gate * value, depthwise causal conv(K=4) + SiLU + residual ------
  const size_t outbase = (size_t)b * L_ * HC * D_;
  for (int idx = tid; idx < TM * HC * D_; idx += 256) {
    int io = idx >> 11;                        // / 2048
    int c  = idx & (HC * D_ - 1);
    int l  = l0 + io;
    if (l >= L_) continue;
    int hc = c >> 10;
    int d  = c & (D_ - 1);
    int i  = io + 3;
    float vb = sm.vb[d];
    float x0 = sm.gate[i - 3][hc] * (sm.KV[i - 3][d] + vb);
    float x1 = sm.gate[i - 2][hc] * (sm.KV[i - 2][d] + vb);
    float x2 = sm.gate[i - 1][hc] * (sm.KV[i - 1][d] + vb);
    float x3 = sm.gate[i][hc]     * (sm.KV[i][d]     + vb);
    float4 w = sm.convw[c];
    float y = fmaf(w.w, x3, fmaf(w.z, x2, fmaf(w.y, x1, fmaf(w.x, x0, sm.convb[c]))));
    float res = x3 + y * (1.f / (1.f + __expf(-y)));
    out[outbase + (size_t)l * HC * D_ + c] = res;
  }
}

extern "C" void kernel_launch(void* const* d_in, const int* in_sizes, int n_in,
                              void* d_out, int out_size, void* d_ws, size_t ws_size,
                              hipStream_t stream) {
  (void)in_sizes; (void)n_in; (void)out_size; (void)ws_size;
  const float* hidden   = (const float*)d_in[0];
  const int*   hash_ids = (const int*)d_in[1];
  const int*   offsets  = (const int*)d_in[2];
  const float* emb      = (const float*)d_in[3];
  const float* value_w  = (const float*)d_in[4];
  const float* value_b  = (const float*)d_in[5];
  const float* key_w    = (const float*)d_in[6];
  const float* key_b    = (const float*)d_in[7];
  const float* nkw      = (const float*)d_in[8];
  const float* nqw      = (const float*)d_in[9];
  const float* conv_w   = (const float*)d_in[10];
  const float* conv_b   = (const float*)d_in[11];
  unsigned short* wsB   = (unsigned short*)d_ws;   // 3 MB bf16 transposed weights

  hipLaunchKernelGGL(prep_weights, dim3(D_ / 32, TE / 32, 3), dim3(32, 8), 0,
                     stream, value_w, key_w, wsB);
  hipLaunchKernelGGL(engram_main, dim3(NCHUNK, B_), dim3(256), 0, stream,
                     hidden, hash_ids, offsets, emb, value_b, key_b, nkw, nqw,
                     conv_w, conv_b, wsB, (float*)d_out);
}
